// Zencoder_70428873720377
// MI455X (gfx1250) — matrix operand order repaired and checked by reading, not compiled, not run on hardware
//
#include <hip/hip_runtime.h>
#include <hip/hip_bf16.h>
#include <math.h>

// ---------------------------------------------------------------------------
// CDNA5 (gfx1250, wave32) implicit-GEMM conv backbone + masked style pooling.
// Matrix path: v_wmma_f32_16x16x32_f16 (f16 A/B, f32 accumulate).
// Round 5: 2x2 register tiling (32 oc x 32 px per wave) -- B staged once per
//          32 output channels (halves L2 traffic of the dominant conv4),
//          8 WMMA per barrier round per wave, B fragments reused across the
//          two M tiles.
// ---------------------------------------------------------------------------

typedef __attribute__((ext_vector_type(16))) _Float16 v16h;
typedef __attribute__((ext_vector_type(8)))  _Float16 h8;
typedef __attribute__((ext_vector_type(8)))  float    v8f;
typedef __attribute__((ext_vector_type(4)))  float    v4f;

#define WAVES      8
#define MT         2                  // M register tiles per block (32 oc)
#define NT         2                  // N register tiles per wave
#define WAVE_PIX   (NT * 16)          // 32 pixels per wave
#define BLOCK_PIX  (WAVES * WAVE_PIX) // 256 pixels per block
#define KC         64                 // K elements staged per barrier round

__device__ __forceinline__ int reflect1(int i, int n) {
  i = (i < 0) ? -i : i;               // pad == 1 -> single reflection suffices
  return (i >= n) ? (2 * n - 2 - i) : i;
}
__device__ __forceinline__ int clampi(int v, int lo, int hi) {
  return v < lo ? lo : (v > hi ? hi : v);
}

enum ConvMode { MODE_ZERO = 0, MODE_REFLECT = 1, MODE_TRANS = 2 };

// ---------------------------------------------------------------------------
// Implicit-GEMM conv: per block  D[32 oc x 256 px] += A[32 x KC] * B[KC x 256]
// K index ordering: k = r*CIN + ci  (r = ky*3+kx kernel tap, ci = in channel).
// grid = (HWout/256, OC/32, B), block = 256 (8 waves, 2x2 WMMA tiles each).
// ---------------------------------------------------------------------------
template<int MODE, int STRIDE, bool TANH_ACT, int CIN>
__global__ __launch_bounds__(256)
void conv_igemm_wmma(const float* __restrict__ in, const float* __restrict__ wgt,
                     const float* __restrict__ bias, float* __restrict__ out,
                     int OC, int Hin, int Win, int Hout, int Wout)
{
  constexpr int Ktot    = CIN * 9;
  constexpr int KSTAGE  = (Ktot + KC - 1) / KC;
  constexpr bool KEXACT = (Ktot % KC) == 0;      // no tail masking needed

  __shared__ _Float16 As[32 * KC];               // weights tile  [m][k] (32 oc)
  __shared__ _Float16 Bs[WAVES][WAVE_PIX * KC];  // im2col tile   [n][k] per wave

  const int tid  = threadIdx.x;
  const int wid  = tid >> 5;
  const int lane = tid & 31;
  const int hi   = lane >> 4;      // lane half (WMMA layout selector)
  const int lo   = lane & 15;

  const int b      = blockIdx.z;
  const int oc0    = blockIdx.y * 32;
  const int pix0   = blockIdx.x * BLOCK_PIX + wid * WAVE_PIX;
  const int HWout  = Hout * Wout;
  const int HinWin = Hin * Win;

  const float* inb = in + (size_t)b * CIN * HinWin;

  v8f acc[MT][NT] = {};

  // staging pixel for this lane (pixel-major B staging)
  const int pn = pix0 + lane;
  const int oy = pn / Wout;
  const int ox = pn - oy * Wout;

  for (int ks = 0; ks < KSTAGE; ++ks) {
    const int k0 = ks * KC;

    // ---- stage A (weights): 32 oc x 64 k, wave 0 (one m-row per lane) ----
    if (wid == 0) {
      const int m   = lane;
      const int ocg = oc0 + m;
      #pragma unroll
      for (int g = 0; g < 4; ++g) {
        float tmp[16];
        #pragma unroll
        for (int e = 0; e < 16; ++e) {
          const int k   = k0 + g * 16 + e;
          const int kcl = (KEXACT || k < Ktot) ? k : (Ktot - 1);
          const int ci  = kcl % CIN;
          const int r   = kcl / CIN;
          if (MODE == MODE_TRANS) {
            // ConvTranspose weight layout [Cin][OC][3][3]
            tmp[e] = wgt[((size_t)ci * OC + ocg) * 9 + r];
          } else {
            // OIHW: element (oc, ci, r)
            tmp[e] = wgt[(size_t)ocg * Ktot + ci * 9 + r];
          }
        }
        h8 p0, p1;
        #pragma unroll
        for (int e = 0; e < 8; ++e) {
          const int ka = k0 + g * 16 + e;
          const int kb = ka + 8;
          p0[e] = (_Float16)((KEXACT || ka < Ktot) ? tmp[e]     : 0.f);
          p1[e] = (_Float16)((KEXACT || kb < Ktot) ? tmp[e + 8] : 0.f);
        }
        *(h8*)&As[m * KC + g * 16]     = p0;
        *(h8*)&As[m * KC + g * 16 + 8] = p1;
      }
    }

    // ---- stage B (im2col gather): lane owns one pixel; two-phase x4 ----
    #pragma unroll
    for (int g = 0; g < 4; ++g) {
      float tmp[16];
      bool  vld[16];
      #pragma unroll
      for (int e = 0; e < 16; ++e) {
        const int k    = k0 + g * 16 + e;
        const bool kval = KEXACT || (k < Ktot);
        const int kcl  = kval ? k : 0;
        const int ci   = kcl % CIN;          // shift/mask for pow2 CIN
        const int r    = kcl / CIN;          // uniform within group for CIN>=16
        const int ky   = r / 3, kx = r - ky * 3;
        bool valid = kval;
        int iy, ix;
        if (MODE == MODE_REFLECT) {
          iy = reflect1(oy - 1 + ky, Hin);
          ix = reflect1(ox - 1 + kx, Win);
        } else if (MODE == MODE_ZERO) {
          const int ty = oy * STRIDE - 1 + ky;
          const int tx = ox * STRIDE - 1 + kx;
          valid = valid && (ty >= 0) && (ty < Hin) && (tx >= 0) && (tx < Win);
          iy = clampi(ty, 0, Hin - 1);
          ix = clampi(tx, 0, Win - 1);
        } else { // MODE_TRANS: iy = (oy + 1 - ky)/2 when even & in range
          const int ty = oy + 1 - ky;
          const int tx = ox + 1 - kx;
          valid = valid && (ty >= 0) && (tx >= 0) && !(ty & 1) && !(tx & 1) &&
                  ((ty >> 1) < Hin) && ((tx >> 1) < Win);
          iy = clampi(ty >> 1, 0, Hin - 1);
          ix = clampi(tx >> 1, 0, Win - 1);
        }
        vld[e] = valid;
        tmp[e] = inb[(size_t)ci * HinWin + iy * Win + ix];
      }
      h8 p0, p1;
      #pragma unroll
      for (int e = 0; e < 8; ++e) {
        p0[e] = (_Float16)(vld[e]     ? tmp[e]     : 0.f);
        p1[e] = (_Float16)(vld[e + 8] ? tmp[e + 8] : 0.f);
      }
      *(h8*)&Bs[wid][lane * KC + g * 16]     = p0;
      *(h8*)&Bs[wid][lane * KC + g * 16 + 8] = p1;
    }

    __syncthreads();

    // ---- 2 k-slabs x (2 M x 2 N) = 8 WMMA per stage; B frags reused ----
    #pragma unroll
    for (int s = 0; s < 2; ++s) {
      const int kbase = s * 32;
      // B 32x16 f16: N = lane&15; VGPR j holds K = hi*16 + 2j, 2j+1
      v16h bfrag[NT];
      #pragma unroll
      for (int t = 0; t < NT; ++t) {
        #pragma unroll
        for (int j = 0; j < 8; ++j) {
          const int kb = kbase + hi * 16 + 2 * j;
          bfrag[t][2 * j]     = Bs[wid][(t * 16 + lo) * KC + kb];
          bfrag[t][2 * j + 1] = Bs[wid][(t * 16 + lo) * KC + kb + 1];
        }
      }
      #pragma unroll
      for (int mt = 0; mt < MT; ++mt) {
        // A 16x32 f16: M = lane&15; hi=0 -> K {0..15}, hi=1 -> K {16..31}
        v16h afrag;
        #pragma unroll
        for (int j = 0; j < 8; ++j) {
          const int ka = kbase + hi * 16 + 2 * j;
          afrag[2 * j]     = As[(mt * 16 + lo) * KC + ka];
          afrag[2 * j + 1] = As[(mt * 16 + lo) * KC + ka + 1];
        }
        #pragma unroll
        for (int t = 0; t < NT; ++t) {
          acc[mt][t] = __builtin_amdgcn_wmma_f32_16x16x32_f16(
              /*neg_a=*/false, afrag, /*neg_b=*/false, bfrag[t],
              /*c_mod=*/(short)0, acc[mt][t], /*reuse_a=*/false, /*reuse_b=*/false);
        }
      }
    }

    __syncthreads();
  }

  // ---- epilogue: C/D layout M = r + 8*hi, N = lane&15 ----
  #pragma unroll
  for (int mt = 0; mt < MT; ++mt) {
    #pragma unroll
    for (int t = 0; t < NT; ++t) {
      #pragma unroll
      for (int r = 0; r < 8; ++r) {
        const int m  = mt * 16 + r + hi * 8;
        const int oc = oc0 + m;
        const int p  = pix0 + t * 16 + lo;
        float v = acc[mt][t][r] + bias[oc];
        if (TANH_ACT) v = tanhf(v);
        out[((size_t)b * OC + oc) * HWout + p] = v;
      }
    }
  }
}

// ---------------------------------------------------------------------------
// InstanceNorm2d (affine=False, biased var) + LeakyReLU(0.2), in place.
// grid = B*C blocks, block = 256. HW is a multiple of 1024 -> float4 clean.
// ---------------------------------------------------------------------------
__global__ __launch_bounds__(256)
void inorm_lrelu_kernel(float* __restrict__ x, int HW)
{
  __shared__ float s_sum[256];
  __shared__ float s_sq[256];
  float* p = x + (size_t)blockIdx.x * HW;
  const int tid = threadIdx.x;
  const int HW4 = HW >> 2;
  v4f* p4 = (v4f*)p;

  float s = 0.f, q = 0.f;
  for (int i = tid; i < HW4; i += 256) {
    const v4f v = p4[i];
    s += v.x + v.y + v.z + v.w;
    q += v.x * v.x + v.y * v.y + v.z * v.z + v.w * v.w;
  }
  s_sum[tid] = s; s_sq[tid] = q;
  __syncthreads();
  for (int off = 128; off > 0; off >>= 1) {
    if (tid < off) { s_sum[tid] += s_sum[tid + off]; s_sq[tid] += s_sq[tid + off]; }
    __syncthreads();
  }
  const float inv  = 1.f / (float)HW;
  const float mean = s_sum[0] * inv;
  const float var  = s_sq[0] * inv - mean * mean;
  const float rstd = rsqrtf(var + 1e-5f);

  for (int i = tid; i < HW4; i += 256) {
    v4f v = p4[i];
    #pragma unroll
    for (int c = 0; c < 4; ++c) {
      const float t = (v[c] - mean) * rstd;
      v[c] = (t >= 0.f) ? t : 0.2f * t;
    }
    p4[i] = v;
  }
}

// ---------------------------------------------------------------------------
// Style codes: nearest-downsample seg 256->128, masked mean over pixels.
// grid = B*19 blocks, block = 256. x: [B,512,128,128], out: [B,19,512].
// ---------------------------------------------------------------------------
__global__ __launch_bounds__(256)
void style_codes_kernel(const float* __restrict__ x, const float* __restrict__ seg,
                        float* __restrict__ out)
{
  constexpr int L = 19, C = 512, Hs = 256, Ws = 256, H = 128, W = 128, HW = H * W;
  __shared__ unsigned char mask[HW];
  __shared__ int cnt_s;

  const int bl = blockIdx.x;
  const int b  = bl / L;
  const int l  = bl - b * L;
  const int tid = threadIdx.x;

  if (tid == 0) cnt_s = 0;
  __syncthreads();

  int cnt = 0;
  for (int p = tid; p < HW; p += 256) {
    const int h = p >> 7, w = p & (W - 1);
    // nearest: src = (dst * 256) / 128 = 2*dst
    const float sv = seg[(((size_t)b * L + l) * Hs + 2 * h) * Ws + 2 * w];
    const unsigned char m = (sv != 0.f) ? 1 : 0;
    mask[p] = m;
    cnt += m;
  }
  atomicAdd(&cnt_s, cnt);
  __syncthreads();

  const int count = cnt_s;
  const float invc = (count > 0) ? (1.f / (float)count) : 0.f;

  for (int c = tid; c < C; c += 256) {
    const v4f* xp = (const v4f*)(x + ((size_t)b * C + c) * HW);
    float s = 0.f;
    for (int p4 = 0; p4 < (HW >> 2); ++p4) {
      const v4f v = xp[p4];
      const int p = p4 << 2;
      s += (mask[p]     ? v.x : 0.f) + (mask[p + 1] ? v.y : 0.f) +
           (mask[p + 2] ? v.z : 0.f) + (mask[p + 3] ? v.w : 0.f);
    }
    out[((size_t)b * L + l) * C + c] = (count > 0) ? s * invc : 0.f;
  }
}

// ---------------------------------------------------------------------------
// Launch sequence
// ---------------------------------------------------------------------------
extern "C" void kernel_launch(void* const* d_in, const int* in_sizes, int n_in,
                              void* d_out, int out_size, void* d_ws, size_t ws_size,
                              hipStream_t stream) {
  (void)in_sizes; (void)n_in; (void)out_size; (void)ws_size;

  const float* image = (const float*)d_in[0];   // [8,3,256,256]
  const float* seg   = (const float*)d_in[1];   // [8,19,256,256]
  const float* w1 = (const float*)d_in[2];  const float* b1 = (const float*)d_in[3];
  const float* w2 = (const float*)d_in[4];  const float* b2 = (const float*)d_in[5];
  const float* w3 = (const float*)d_in[6];  const float* b3 = (const float*)d_in[7];
  const float* wt = (const float*)d_in[8];  const float* bt = (const float*)d_in[9];
  const float* w4 = (const float*)d_in[10]; const float* b4 = (const float*)d_in[11];
  float* out = (float*)d_out;                   // [8,19,512]

  // workspace layout (floats)
  float* ws = (float*)d_ws;
  const size_t n1 = (size_t)8 * 32  * 256 * 256;  // x1
  const size_t n2 = (size_t)8 * 64  * 128 * 128;  // x2
  const size_t n3 = (size_t)8 * 128 * 64  * 64;   // x3
  const size_t n4 = (size_t)8 * 256 * 128 * 128;  // x4
  float* x1 = ws;
  float* x2 = x1 + n1;
  float* x3 = x2 + n2;
  float* x4 = x3 + n3;
  float* x5 = x4 + n4;                            // [8,512,128,128]

  const dim3 blk(256);

  // conv1: reflect-pad 3->32, 256x256
  conv_igemm_wmma<MODE_REFLECT, 1, false, 3><<<dim3(256, 1, 8), blk, 0, stream>>>(
      image, w1, b1, x1, 32, 256, 256, 256, 256);
  inorm_lrelu_kernel<<<dim3(8 * 32), blk, 0, stream>>>(x1, 256 * 256);

  // conv2: zero-pad stride2 32->64, 256 -> 128
  conv_igemm_wmma<MODE_ZERO, 2, false, 32><<<dim3(64, 2, 8), blk, 0, stream>>>(
      x1, w2, b2, x2, 64, 256, 256, 128, 128);
  inorm_lrelu_kernel<<<dim3(8 * 64), blk, 0, stream>>>(x2, 128 * 128);

  // conv3: zero-pad stride2 64->128, 128 -> 64
  conv_igemm_wmma<MODE_ZERO, 2, false, 64><<<dim3(16, 4, 8), blk, 0, stream>>>(
      x2, w3, b3, x3, 128, 128, 128, 64, 64);
  inorm_lrelu_kernel<<<dim3(8 * 128), blk, 0, stream>>>(x3, 64 * 64);

  // convT: 128->256, 64 -> 128 (stride2, pad1, outpad1)
  conv_igemm_wmma<MODE_TRANS, 2, false, 128><<<dim3(64, 8, 8), blk, 0, stream>>>(
      x3, wt, bt, x4, 256, 64, 64, 128, 128);
  inorm_lrelu_kernel<<<dim3(8 * 256), blk, 0, stream>>>(x4, 128 * 128);

  // conv4: reflect-pad 256->512, 128x128, fused tanh
  conv_igemm_wmma<MODE_REFLECT, 1, true, 256><<<dim3(64, 16, 8), blk, 0, stream>>>(
      x4, w4, b4, x5, 512, 128, 128, 128, 128);

  // masked style pooling -> [8,19,512]
  style_codes_kernel<<<dim3(8 * 19), blk, 0, stream>>>(x5, seg, out);
}
